// MultiHeadAttention_4707284156682
// MI455X (gfx1250) — compile-verified, matrix operand order checked by reading, not run on hardware
//
#include <hip/hip_runtime.h>

// MI455X / gfx1250 fused causal multi-head attention forward.
// All matmuls via v_wmma_f32_16x16x32_f16 (wave32 WMMA), flash-style fused
// softmax (no [T,T] score materialization), f32 accumulate everywhere.
// Register-blocked: GEMM waves compute 32x64 tiles (1.5 loads/WMMA),
// attention waves process 32 q-rows reusing K/V fragments (1.0 loads/WMMA).

typedef _Float16 half_t;
typedef __attribute__((ext_vector_type(16))) _Float16 v16h;
typedef __attribute__((ext_vector_type(8)))  _Float16 v8h;
typedef __attribute__((ext_vector_type(8)))  float    v8f;

#define N_EMBD    384
#define NUM_HEADS 6
#define HEAD_SIZE 64
#define BATCH     32
#define SEQ       1024

// ---------------------------------------------------------------------------
// CDNA5 16-bit 16x32 A/B fragment loader (ISA 7.12.2):
//   lane L holds row (A) / column (B) = row0 + (L&15)
//   elements 0..7  : K = k0 +      (L>>4)*8 + j   (contiguous 16B load)
//   elements 8..15 : K = k0 + 16 + (L>>4)*8 + j   (contiguous 16B load)
// ---------------------------------------------------------------------------
static __device__ __forceinline__ v16h wmma_frag_load(const half_t* __restrict__ base,
                                                      int row0, int k0, int ld)
{
    int lane = threadIdx.x & 31;
    const half_t* p = base + (size_t)(row0 + (lane & 15)) * ld + k0 + ((lane >> 4) << 3);
    v8h lo = *(const v8h*)p;
    v8h hi = *(const v8h*)(p + 16);
    return __builtin_shufflevector(lo, hi, 0,1,2,3,4,5,6,7,8,9,10,11,12,13,14,15);
}

static __device__ __forceinline__ v8f wmma_f16(v16h a, v16h b, v8f c)
{
    // 8 args: (neg_a, A, neg_b, B, c_mod, C, reuse_a, reuse_b)
    return __builtin_amdgcn_wmma_f32_16x16x32_f16(false, a, false, b, (short)0, c, false, false);
}

// reductions across the 16-lane half-group that shares a C-matrix row
static __device__ __forceinline__ float redmax16(float v)
{
    v = fmaxf(v, __shfl_xor(v, 1, 32));
    v = fmaxf(v, __shfl_xor(v, 2, 32));
    v = fmaxf(v, __shfl_xor(v, 4, 32));
    v = fmaxf(v, __shfl_xor(v, 8, 32));
    return v;
}
static __device__ __forceinline__ float redsum16(float v)
{
    v += __shfl_xor(v, 1, 32);
    v += __shfl_xor(v, 2, 32);
    v += __shfl_xor(v, 4, 32);
    v += __shfl_xor(v, 8, 32);
    return v;
}

// ---------------------------------------------------------------------------
// Conversion kernels (fp32 -> f16, plus weight transpose so B-fragments get
// contiguous-K 128-bit loads).
// ---------------------------------------------------------------------------
__global__ void cvt_f32_to_f16(const float* __restrict__ in, half_t* __restrict__ out, int n)
{
    int i = blockIdx.x * blockDim.x + threadIdx.x;
    int stride = gridDim.x * blockDim.x;
    for (; i < n; i += stride) out[i] = (half_t)in[i];
}

__global__ void cvt_transpose_f16(const float* __restrict__ W, half_t* __restrict__ WT,
                                  int K, int N)   // W:[K,N] -> WT:[N,K]
{
    int i = blockIdx.x * blockDim.x + threadIdx.x;
    if (i < K * N) {
        int kk = i / N, n = i % N;
        WT[(size_t)n * K + kk] = (half_t)W[i];
    }
}

// ---------------------------------------------------------------------------
// QKV GEMM: [32768,384] x [384,1152].
// One wave computes a 32x64 tile (2 M-subtiles x 4 N-subtiles, 8 WMMAs/K-step).
// A 64-wide N tile lies entirely within one of {q,k,v} and one head, so the
// scatter epilogue (q,k row-major [B,H,T,D]; v transposed [B,H,D,T]) is
// wave-uniform.
// ---------------------------------------------------------------------------
__global__ void __launch_bounds__(256) qkv_gemm_kernel(const half_t* __restrict__ xh,
                                                       const half_t* __restrict__ WaT,
                                                       half_t* __restrict__ qd,
                                                       half_t* __restrict__ kd,
                                                       half_t* __restrict__ vT)
{
    int gw = (blockIdx.x * blockDim.x + threadIdx.x) >> 5;
    const int MT = (BATCH * SEQ) / 32;              // 1024
    int mt = gw % MT;
    int nt = gw / MT;                               // 0..17
    if (nt >= (3 * N_EMBD) / 64) return;
    int m0 = mt * 32, n0 = nt * 64;

    v8f acc[2][4] = {};
    #pragma unroll
    for (int k0 = 0; k0 < N_EMBD; k0 += 32) {
        v16h a0 = wmma_frag_load(xh, m0,      k0, N_EMBD);
        v16h a1 = wmma_frag_load(xh, m0 + 16, k0, N_EMBD);
        v16h b0 = wmma_frag_load(WaT, n0,      k0, N_EMBD);
        v16h b1 = wmma_frag_load(WaT, n0 + 16, k0, N_EMBD);
        v16h b2 = wmma_frag_load(WaT, n0 + 32, k0, N_EMBD);
        v16h b3 = wmma_frag_load(WaT, n0 + 48, k0, N_EMBD);
        acc[0][0] = wmma_f16(a0, b0, acc[0][0]);
        acc[0][1] = wmma_f16(a0, b1, acc[0][1]);
        acc[0][2] = wmma_f16(a0, b2, acc[0][2]);
        acc[0][3] = wmma_f16(a0, b3, acc[0][3]);
        acc[1][0] = wmma_f16(a1, b0, acc[1][0]);
        acc[1][1] = wmma_f16(a1, b1, acc[1][1]);
        acc[1][2] = wmma_f16(a1, b2, acc[1][2]);
        acc[1][3] = wmma_f16(a1, b3, acc[1][3]);
    }

    int lane = threadIdx.x & 31;
    int nl = lane & 15;
    int hs = lane >> 4;
    int seg = n0 / N_EMBD;                          // 0=q, 1=k, 2=v (wave-uniform)
    int h = (n0 - seg * N_EMBD) >> 6;               // head (wave-uniform)
    #pragma unroll
    for (int i = 0; i < 2; ++i) {
        #pragma unroll
        for (int r = 0; r < 8; ++r) {
            int row = m0 + 16 * i + r + 8 * hs;     // global token index
            int b_ = row >> 10, t = row & 1023;
            size_t rowbase = ((size_t)b_ * NUM_HEADS + h) * SEQ;
            #pragma unroll
            for (int j = 0; j < 4; ++j) {
                int d = 16 * j + nl;
                half_t val = (half_t)acc[i][j][r];
                if (seg == 0)      qd[(rowbase + t) * HEAD_SIZE + d] = val;
                else if (seg == 1) kd[(rowbase + t) * HEAD_SIZE + d] = val;
                else               vT[(((size_t)b_ * NUM_HEADS + h) * HEAD_SIZE + d) * SEQ + t] = val;
            }
        }
    }
}

// ---------------------------------------------------------------------------
// Flash attention: one wave per (b, h, 32-row q block). Running max/sum + O
// accumulators (2 x 16x64 f32) in VGPRs; K/V fragments reused across both
// q-subtiles: 16 b128 loads feed 16 WMMAs per 32-key step.
// P tiles converted C-layout -> A-layout via a 2KB per-wave LDS bounce.
// ---------------------------------------------------------------------------
__global__ void __launch_bounds__(256) attn_kernel(const half_t* __restrict__ qd,
                                                   const half_t* __restrict__ kd,
                                                   const half_t* __restrict__ vT,
                                                   half_t* __restrict__ yh)
{
    __shared__ __align__(16) _Float16 plds[8][2][16 * 32];  // per-wave P tiles
    int wib  = threadIdx.x >> 5;
    int lane = threadIdx.x & 31;
    int gw = blockIdx.x * 8 + wib;
    int qt = gw & (SEQ / 32 - 1);                   // 0..31
    int bh = gw >> 5;                               // 0..191
    int b_ = bh / NUM_HEADS, h = bh % NUM_HEADS;

    const half_t* qp = qd + (size_t)bh * SEQ * HEAD_SIZE;
    const half_t* kp = kd + (size_t)bh * SEQ * HEAD_SIZE;
    const half_t* vp = vT + (size_t)bh * HEAD_SIZE * SEQ;

    int q0 = qt * 32;
    int hs = lane >> 4;
    int nl = lane & 15;

    v16h aq[2][2];
    aq[0][0] = wmma_frag_load(qp, q0,      0,  HEAD_SIZE);
    aq[0][1] = wmma_frag_load(qp, q0,      32, HEAD_SIZE);
    aq[1][0] = wmma_frag_load(qp, q0 + 16, 0,  HEAD_SIZE);
    aq[1][1] = wmma_frag_load(qp, q0 + 16, 32, HEAD_SIZE);

    float mrow[2][8], lrow[2][8];
    v8f o[2][4] = {};
    #pragma unroll
    for (int i = 0; i < 2; ++i)
        #pragma unroll
        for (int r = 0; r < 8; ++r) { mrow[i][r] = -__builtin_inff(); lrow[i][r] = 0.f; }

    const float scale = 0.125f;                     // 1/sqrt(64)
    int nsteps = qt + 1;                            // keys [0, q0+32) in steps of 32
    for (int s = 0; s < nsteps; ++s) {
        int k0 = s * 32;
        // K fragments shared by both q-subtiles
        v16h bk0a = wmma_frag_load(kp, k0,      0,  HEAD_SIZE);
        v16h bk0b = wmma_frag_load(kp, k0,      32, HEAD_SIZE);
        v16h bk1a = wmma_frag_load(kp, k0 + 16, 0,  HEAD_SIZE);
        v16h bk1b = wmma_frag_load(kp, k0 + 16, 32, HEAD_SIZE);

        v8f c[2][2] = {};
        #pragma unroll
        for (int i = 0; i < 2; ++i) {
            c[i][0] = wmma_f16(aq[i][0], bk0a, c[i][0]);
            c[i][0] = wmma_f16(aq[i][1], bk0b, c[i][0]);
            c[i][1] = wmma_f16(aq[i][0], bk1a, c[i][1]);
            c[i][1] = wmma_f16(aq[i][1], bk1b, c[i][1]);
        }

        // online softmax update; C-layout: row = r + 8*hs, col = nl
        #pragma unroll
        for (int i = 0; i < 2; ++i) {
            #pragma unroll
            for (int r = 0; r < 8; ++r) {
                int rowg = q0 + 16 * i + r + 8 * hs;
                float s0 = (k0 + nl      <= rowg) ? c[i][0][r] * scale : -__builtin_inff();
                float s1 = (k0 + 16 + nl <= rowg) ? c[i][1][r] * scale : -__builtin_inff();
                float mnew  = fmaxf(mrow[i][r], redmax16(fmaxf(s0, s1)));
                float alpha = __expf(mrow[i][r] - mnew);
                float p0 = __expf(s0 - mnew);
                float p1 = __expf(s1 - mnew);
                lrow[i][r] = lrow[i][r] * alpha + redsum16(p0 + p1);
                mrow[i][r] = mnew;
                o[i][0][r] *= alpha; o[i][1][r] *= alpha;
                o[i][2][r] *= alpha; o[i][3][r] *= alpha;
                int rl = r + 8 * hs;
                plds[wib][i][rl * 32 + nl]      = (_Float16)p0;
                plds[wib][i][rl * 32 + 16 + nl] = (_Float16)p1;
            }
        }

        // V fragments shared by both q-subtiles (v^T rows contiguous over keys)
        v16h bv0 = wmma_frag_load(vp, 0,  k0, SEQ);
        v16h bv1 = wmma_frag_load(vp, 16, k0, SEQ);
        v16h bv2 = wmma_frag_load(vp, 32, k0, SEQ);
        v16h bv3 = wmma_frag_load(vp, 48, k0, SEQ);

        #pragma unroll
        for (int i = 0; i < 2; ++i) {
            // reload P in A-fragment layout (same wave: DS ops in order, no barrier)
            const _Float16* pb = &plds[wib][i][0];
            v8h plo = *(const v8h*)(pb + nl * 32 +      hs * 8);
            v8h phi = *(const v8h*)(pb + nl * 32 + 16 + hs * 8);
            v16h aP = __builtin_shufflevector(plo, phi, 0,1,2,3,4,5,6,7,8,9,10,11,12,13,14,15);
            o[i][0] = wmma_f16(aP, bv0, o[i][0]);
            o[i][1] = wmma_f16(aP, bv1, o[i][1]);
            o[i][2] = wmma_f16(aP, bv2, o[i][2]);
            o[i][3] = wmma_f16(aP, bv3, o[i][3]);
        }
    }

    // normalize and write y in [B, T, H*D] so projection A-fragments are contiguous
    #pragma unroll
    for (int i = 0; i < 2; ++i) {
        #pragma unroll
        for (int r = 0; r < 8; ++r) {
            float inv = 1.0f / lrow[i][r];
            int t = q0 + 16 * i + r + 8 * hs;
            size_t base = ((size_t)b_ * SEQ + t) * N_EMBD + h * HEAD_SIZE + nl;
            yh[base]      = (half_t)(o[i][0][r] * inv);
            yh[base + 16] = (half_t)(o[i][1][r] * inv);
            yh[base + 32] = (half_t)(o[i][2][r] * inv);
            yh[base + 48] = (half_t)(o[i][3][r] * inv);
        }
    }
}

// ---------------------------------------------------------------------------
// Output projection: [32768,384] x [384,384] + bias -> fp32 out.
// 32x64 tile per wave, same blocking as QKV.
// ---------------------------------------------------------------------------
__global__ void __launch_bounds__(256) proj_gemm_kernel(const half_t* __restrict__ yh,
                                                        const half_t* __restrict__ WpT,
                                                        const float* __restrict__ bias,
                                                        float* __restrict__ out)
{
    int gw = (blockIdx.x * blockDim.x + threadIdx.x) >> 5;
    const int MT = (BATCH * SEQ) / 32;              // 1024
    int mt = gw % MT;
    int nt = gw / MT;                               // 0..5
    if (nt >= N_EMBD / 64) return;
    int m0 = mt * 32, n0 = nt * 64;

    v8f acc[2][4] = {};
    #pragma unroll
    for (int k0 = 0; k0 < N_EMBD; k0 += 32) {
        v16h a0 = wmma_frag_load(yh, m0,      k0, N_EMBD);
        v16h a1 = wmma_frag_load(yh, m0 + 16, k0, N_EMBD);
        v16h b0 = wmma_frag_load(WpT, n0,      k0, N_EMBD);
        v16h b1 = wmma_frag_load(WpT, n0 + 16, k0, N_EMBD);
        v16h b2 = wmma_frag_load(WpT, n0 + 32, k0, N_EMBD);
        v16h b3 = wmma_frag_load(WpT, n0 + 48, k0, N_EMBD);
        acc[0][0] = wmma_f16(a0, b0, acc[0][0]);
        acc[0][1] = wmma_f16(a0, b1, acc[0][1]);
        acc[0][2] = wmma_f16(a0, b2, acc[0][2]);
        acc[0][3] = wmma_f16(a0, b3, acc[0][3]);
        acc[1][0] = wmma_f16(a1, b0, acc[1][0]);
        acc[1][1] = wmma_f16(a1, b1, acc[1][1]);
        acc[1][2] = wmma_f16(a1, b2, acc[1][2]);
        acc[1][3] = wmma_f16(a1, b3, acc[1][3]);
    }

    int lane = threadIdx.x & 31;
    int nl = lane & 15;
    int hs = lane >> 4;
    float bv[4];
    #pragma unroll
    for (int j = 0; j < 4; ++j) bv[j] = bias[n0 + 16 * j + nl];
    #pragma unroll
    for (int i = 0; i < 2; ++i) {
        #pragma unroll
        for (int r = 0; r < 8; ++r) {
            int row = m0 + 16 * i + r + 8 * hs;
            #pragma unroll
            for (int j = 0; j < 4; ++j)
                out[(size_t)row * N_EMBD + n0 + 16 * j + nl] = acc[i][j][r] + bv[j];
        }
    }
}

// ---------------------------------------------------------------------------
extern "C" void kernel_launch(void* const* d_in, const int* in_sizes, int n_in,
                              void* d_out, int out_size, void* d_ws, size_t ws_size,
                              hipStream_t stream)
{
    const float* x      = (const float*)d_in[0];
    const float* W_attn = (const float*)d_in[1];
    const float* W_proj = (const float*)d_in[2];
    const float* b_proj = (const float*)d_in[3];
    float* out = (float*)d_out;

    const size_t MBT     = (size_t)BATCH * SEQ;                 // 32768 tokens
    const size_t bytes_a = MBT * N_EMBD * sizeof(half_t);       // 25.2 MB each

    char* w = (char*)d_ws;
    half_t* xh  = (half_t*)w; w += bytes_a;
    half_t* WaT = (half_t*)w; w += (size_t)3 * N_EMBD * N_EMBD * sizeof(half_t);
    half_t* WpT = (half_t*)w; w += (size_t)N_EMBD * N_EMBD * sizeof(half_t);
    half_t* qd  = (half_t*)w; w += bytes_a;
    half_t* kd  = (half_t*)w; w += bytes_a;
    half_t* vT  = (half_t*)w; w += bytes_a;
    half_t* yh  = (half_t*)w; w += bytes_a;

    {   // x -> f16
        int n = (int)(MBT * N_EMBD);
        cvt_f32_to_f16<<<4096, 256, 0, stream>>>(x, xh, n);
    }
    {   // W_attn^T, W_proj^T -> f16
        int n = 3 * N_EMBD * N_EMBD;
        cvt_transpose_f16<<<(n + 255) / 256, 256, 0, stream>>>(W_attn, WaT, N_EMBD, 3 * N_EMBD);
        n = N_EMBD * N_EMBD;
        cvt_transpose_f16<<<(n + 255) / 256, 256, 0, stream>>>(W_proj, WpT, N_EMBD, N_EMBD);
    }
    {   // QKV GEMM: 1024 M-tiles x 18 N-tiles
        int waves = ((BATCH * SEQ) / 32) * ((3 * N_EMBD) / 64); // 18432
        qkv_gemm_kernel<<<(waves * 32 + 255) / 256, 256, 0, stream>>>(xh, WaT, qd, kd, vT);
    }
    {   // flash attention: one wave per (b,h,32-row q block)
        int waves = BATCH * NUM_HEADS * (SEQ / 32);             // 6144
        attn_kernel<<<waves / 8, 256, 0, stream>>>(qd, kd, vT, yh);
    }
    {   // output projection + bias: 1024 x 6 tiles
        int waves = ((BATCH * SEQ) / 32) * (N_EMBD / 64);       // 6144
        proj_gemm_kernel<<<(waves * 32 + 255) / 256, 256, 0, stream>>>(yh, WpT, b_proj, out);
    }
}
